// CasualSelfAttention_56667798504030
// MI455X (gfx1250) — matrix operand order repaired and checked by reading, not compiled, not run on hardware
//
#include <hip/hip_runtime.h>
#include <hip/hip_bf16.h>

// ---------------------------------------------------------------------------
// Causal self-attention for MI455X (gfx1250, wave32, WMMA bf16).
// Phase 0: one-shot f32 -> bf16 conversion of x and the live rows of w.
// Phase 1: QKV GEMM (3E live columns), bf16 WMMA f32-accum, double-buffered
//          LDS staged with GLOBAL_LOAD_ASYNC_TO_LDS_B128 (ASYNCcnt).
// Phase 2: Flash attention with online softmax, bf16 WMMA for QK^T and PV.
// ---------------------------------------------------------------------------

typedef __attribute__((ext_vector_type(16))) __bf16 v16bf;
typedef __attribute__((ext_vector_type(8)))  float  v8f;
typedef __attribute__((ext_vector_type(4)))  int    v4i;
typedef unsigned short u16;
typedef unsigned int   u32;

union FragB {
  v16bf v;
  uint4 q[2];
};

__device__ __forceinline__ u16 f2bf(float f) {
  u32 u = __float_as_uint(f);
  u32 r = (u + 0x7FFFu + ((u >> 16) & 1u)) >> 16;  // round-to-nearest-even
  return (u16)r;
}
__device__ __forceinline__ u32 pack2(float lo, float hi) {
  return (u32)f2bf(lo) | ((u32)f2bf(hi) << 16);
}
__device__ __forceinline__ v8f wmma_bf16(const FragB& a, const FragB& b, v8f c) {
  return __builtin_amdgcn_wmma_f32_16x16x32_bf16(false, a.v, false, b.v,
                                                 (short)0, c, false, false);
}

// ---- CDNA5 async global->LDS copy (ASYNCcnt) with guarded fallback --------
#if defined(__has_builtin)
#if __has_builtin(__builtin_amdgcn_global_load_async_to_lds_b128)
#define HAVE_ASYNC_LDS 1
#endif
#endif

#ifdef HAVE_ASYNC_LDS
#pragma message("gfx1250: using __builtin_amdgcn_global_load_async_to_lds_b128")
typedef __attribute__((address_space(1))) v4i gas_v4i;  // global int4
typedef __attribute__((address_space(3))) v4i las_v4i;  // LDS int4
__device__ __forceinline__ void async_copy16(const u16* g, u16* l) {
  __builtin_amdgcn_global_load_async_to_lds_b128(
      (gas_v4i*)g, (las_v4i*)l, 0, 0);
}
#if __has_builtin(__builtin_amdgcn_s_wait_asynccnt)
#define WAIT_ASYNC8 __builtin_amdgcn_s_wait_asynccnt(8)
#define WAIT_ASYNC0 __builtin_amdgcn_s_wait_asynccnt(0)
#else
#define WAIT_ASYNC8 asm volatile("s_wait_asynccnt 8" ::: "memory")
#define WAIT_ASYNC0 asm volatile("s_wait_asynccnt 0" ::: "memory")
#endif
#else
#pragma message("gfx1250: async-to-LDS builtin NOT available; sync fallback")
__device__ __forceinline__ void async_copy16(const u16* g, u16* l) {
  *(uint4*)l = *(const uint4*)g;
}
#define WAIT_ASYNC8 do {} while (0)
#define WAIT_ASYNC0 do {} while (0)
#endif

#define B_DIM 4
#define T_DIM 2048
#define E_DIM 1024
#define H_DIM 16
#define D_DIM 64
#define QK_ELEMS (B_DIM * H_DIM * T_DIM * D_DIM)  // 8388608 per q/k/v tensor
#define X_ELEMS  (B_DIM * T_DIM * E_DIM)          // 8388608
#define W_ELEMS  (3 * E_DIM * E_DIM)              // 3145728 (live rows only)

// ===========================================================================
// Kernel 0: bulk f32 -> bf16 (8 elements / thread), pure bandwidth.
// ===========================================================================
__global__ __launch_bounds__(256) void cvt_bf16_kernel(
    const float* __restrict__ src, u16* __restrict__ dst, int n8) {
  const int i = blockIdx.x * 256 + threadIdx.x;
  if (i < n8) {
    const float4* s = (const float4*)src + (size_t)i * 2;
    float4 a = s[0], b = s[1];
    uint4 o;
    o.x = pack2(a.x, a.y);
    o.y = pack2(a.z, a.w);
    o.z = pack2(b.x, b.y);
    o.w = pack2(b.z, b.w);
    ((uint4*)dst)[i] = o;
  }
}

// ===========================================================================
// Kernel 1: qkv = x @ w[0:3E].T + b[0:3E], written as bf16 [B,H,T,D] x3.
// M=8192, N=3072, K=1024. Tile 128x128x64, 8 waves, double-buffered LDS
// staged via async global->LDS copies.
// ===========================================================================
#define GE_BM 128
#define GE_BN 128
#define GE_BK 64
#define GE_LDP (GE_BK + 8)          // 144B row stride: 16B-aligned, no conflicts
#define GE_NSTEP (E_DIM / GE_BK)    // 16

__device__ __forceinline__ void stage_tile(const u16* __restrict__ gA,
                                           const u16* __restrict__ gW,
                                           u16 (*la)[GE_LDP], u16 (*lw)[GE_LDP],
                                           int tid, int k0) {
  // 128x64 bf16 tile = 1024 16B-chunks; 256 threads x 4 chunks each.
#pragma unroll
  for (int r = 0; r < 4; ++r) {
    const int ch  = tid + r * 256;
    const int row = ch >> 3;
    const int c8  = (ch & 7) * 8;
    async_copy16(gA + (size_t)row * E_DIM + k0 + c8, &la[row][c8]);
  }
#pragma unroll
  for (int r = 0; r < 4; ++r) {
    const int ch  = tid + r * 256;
    const int row = ch >> 3;
    const int c8  = (ch & 7) * 8;
    async_copy16(gW + (size_t)row * E_DIM + k0 + c8, &lw[row][c8]);
  }
}

__global__ __launch_bounds__(256) void qkv_gemm_kernel(
    const u16* __restrict__ xb, const u16* __restrict__ wb,
    const float* __restrict__ bias, u16* __restrict__ qkv) {
  __shared__ u16 aLds[2][GE_BM][GE_LDP];
  __shared__ u16 wLds[2][GE_BN][GE_LDP];

  const int tid      = threadIdx.x;
  const int lane     = tid & 31;
  const int wid      = tid >> 5;
  const int lane16   = lane & 15;
  const int laneHalf = lane >> 4;
  const int waveM    = wid >> 2;   // 2 wave-rows (64 M each)
  const int waveN    = wid & 3;    // 4 wave-cols (32 N each)
  const int mBase    = blockIdx.y * GE_BM;
  const int nBase    = blockIdx.x * GE_BN;

  const u16* gA = xb + (size_t)mBase * E_DIM;
  const u16* gW = wb + (size_t)nBase * E_DIM;

  v8f acc[4][2];
#pragma unroll
  for (int mi = 0; mi < 4; ++mi)
#pragma unroll
    for (int ni = 0; ni < 2; ++ni) {
      v8f z = {};
      acc[mi][ni] = z;
    }

  // Prologue: stage first tile pair.
  stage_tile(gA, gW, aLds[0], wLds[0], tid, 0);

  for (int j = 0; j < GE_NSTEP; ++j) {
    const int cur = j & 1;
    if (j + 1 < GE_NSTEP) {
      stage_tile(gA, gW, aLds[1 - cur], wLds[1 - cur], tid, (j + 1) * GE_BK);
      WAIT_ASYNC8;   // 8 newer copies outstanding => current tile complete
    } else {
      WAIT_ASYNC0;
    }
    __syncthreads();

#pragma unroll
    for (int kh = 0; kh < 2; ++kh) {
      const int kb = kh * 32;
      // A fragments (ISA layout: lanes 0-15 K{0-7,16-23}, 16-31 K{8-15,24-31})
      FragB af[4];
#pragma unroll
      for (int mi = 0; mi < 4; ++mi) {
        const int row = waveM * 64 + mi * 16 + lane16;
        af[mi].q[0] = *(const uint4*)&aLds[cur][row][kb + laneHalf * 8];
        af[mi].q[1] = *(const uint4*)&aLds[cur][row][kb + 16 + laneHalf * 8];
      }
      // B fragments: lane = N column, same interleaved K chunks as A
      FragB bf[2];
#pragma unroll
      for (int ni = 0; ni < 2; ++ni) {
        const int col = waveN * 32 + ni * 16 + lane16;
        bf[ni].q[0] = *(const uint4*)&wLds[cur][col][kb + laneHalf * 8];
        bf[ni].q[1] = *(const uint4*)&wLds[cur][col][kb + 16 + laneHalf * 8];
      }
#pragma unroll
      for (int mi = 0; mi < 4; ++mi)
#pragma unroll
        for (int ni = 0; ni < 2; ++ni)
          acc[mi][ni] = wmma_bf16(af[mi], bf[ni], acc[mi][ni]);
    }
    __syncthreads();  // all waves done with buf[cur] before it is re-staged
  }

  // Epilogue: +bias, bf16, scatter into q/k/v [B,H,T,D]
#pragma unroll
  for (int mi = 0; mi < 4; ++mi) {
#pragma unroll
    for (int ni = 0; ni < 2; ++ni) {
      const int ncol  = nBase + waveN * 32 + ni * 16 + lane16;
      const float bv  = bias[ncol];
      const int which = ncol >> 10;        // 0=q 1=k 2=v
      const int nc    = ncol & 1023;
      const int h     = nc >> 6;
      const int d     = nc & 63;
      u16* dst = qkv + (size_t)which * QK_ELEMS;
#pragma unroll
      for (int i = 0; i < 8; ++i) {
        const int m  = mBase + waveM * 64 + mi * 16 + i + laneHalf * 8;
        const int t  = m & (T_DIM - 1);
        const int bb = m >> 11;
        dst[(((size_t)(bb * H_DIM + h)) * T_DIM + t) * D_DIM + d] =
            f2bf(acc[mi][ni][i] + bv);
      }
    }
  }
}

// ===========================================================================
// Kernel 2: causal flash attention. One block = 64 q rows of one (b,h).
// 4 waves x 16 q rows. Key/value blocks of 64, online softmax (exp2 domain).
// ===========================================================================
__global__ __launch_bounds__(128) void flash_attn_kernel(
    const u16* __restrict__ qg, const u16* __restrict__ kg,
    const u16* __restrict__ vg, float* __restrict__ out) {
  __shared__ u16 vT[64][72];        // V^T tile: [d][k], padded
  __shared__ u16 pT[4][16][72];     // per-wave P tile: [row][k], padded

  const int tid      = threadIdx.x;
  const int lane     = tid & 31;
  const int wid      = tid >> 5;
  const int lane16   = lane & 15;
  const int laneHalf = lane >> 4;

  const int bh   = blockIdx.y;         // b*16 + h
  const int qblk = blockIdx.x;
  const int q0   = qblk * 64;

  const u16* qh = qg + (size_t)bh * T_DIM * D_DIM;
  const u16* kh = kg + (size_t)bh * T_DIM * D_DIM;
  const u16* vh = vg + (size_t)bh * T_DIM * D_DIM;

  // Q fragments for this wave's 16 rows (A layout, interleaved K chunks)
  FragB qf[2];
  {
    const int qrow = q0 + wid * 16 + lane16;
#pragma unroll
    for (int dh = 0; dh < 2; ++dh) {
      const u16* p = qh + (size_t)qrow * D_DIM + dh * 32 + laneHalf * 8;
      qf[dh].q[0] = *(const uint4*)p;
      qf[dh].q[1] = *(const uint4*)(p + 16);
    }
  }

  v8f o[4];
#pragma unroll
  for (int dt = 0; dt < 4; ++dt) { v8f z = {}; o[dt] = z; }
  float m_i[8], l_i[8];
#pragma unroll
  for (int i = 0; i < 8; ++i) { m_i[i] = -1e30f; l_i[i] = 0.0f; }

  const float sc = 0.125f * 1.44269504088896f;  // 1/sqrt(D) * log2(e)

  // Transpose-staging map: thread handles 4 consecutive V rows x 8 d-cols.
  const int sr4 = (tid & 15) * 4;   // V row group: 0,4,...,60
  const int sd8 = (tid >> 4) * 8;   // d group: 0,8,...,56

  for (int j = 0; j <= qblk; ++j) {
    const int kb = j * 64;

    // Stage V^T: 4 coalesced b128 row loads, 8 x b64 transposed LDS stores.
    {
      const u16* vp = vh + (size_t)(kb + sr4) * D_DIM + sd8;
      uint4 r0 = *(const uint4*)(vp);
      uint4 r1 = *(const uint4*)(vp + D_DIM);
      uint4 r2 = *(const uint4*)(vp + 2 * D_DIM);
      uint4 r3 = *(const uint4*)(vp + 3 * D_DIM);
      const u16* e0 = (const u16*)&r0;
      const u16* e1 = (const u16*)&r1;
      const u16* e2 = (const u16*)&r2;
      const u16* e3 = (const u16*)&r3;
#pragma unroll
      for (int d = 0; d < 8; ++d) {
        uint2 p;
        p.x = (u32)e0[d] | ((u32)e1[d] << 16);
        p.y = (u32)e2[d] | ((u32)e3[d] << 16);
        *(uint2*)&vT[sd8 + d][sr4] = p;   // 8B-aligned (144*d + 2*r4)
      }
    }
    __syncthreads();

    // S = Q K^T  (4 key sub-tiles of 16, K frags straight from global)
    v8f s[4];
#pragma unroll
    for (int nt = 0; nt < 4; ++nt) { v8f z = {}; s[nt] = z; }
#pragma unroll
    for (int nt = 0; nt < 4; ++nt) {
#pragma unroll
      for (int dh = 0; dh < 2; ++dh) {
        FragB kf;
        const u16* kp = kh + (size_t)(kb + nt * 16 + lane16) * D_DIM +
                        dh * 32 + laneHalf * 8;
        kf.q[0] = *(const uint4*)kp;
        kf.q[1] = *(const uint4*)(kp + 16);
        s[nt] = wmma_bf16(qf[dh], kf, s[nt]);
      }
    }

    // Scale + causal mask + row max (branchless, exp2 domain)
    float rowmax[8];
#pragma unroll
    for (int i = 0; i < 8; ++i) rowmax[i] = -1e30f;
#pragma unroll
    for (int nt = 0; nt < 4; ++nt) {
      const int kcol = kb + nt * 16 + lane16;
#pragma unroll
      for (int i = 0; i < 8; ++i) {
        const int qr = q0 + wid * 16 + i + laneHalf * 8;
        float val = s[nt][i] * sc;
        val = (kcol > qr) ? -1e30f : val;
        s[nt][i] = val;
        rowmax[i] = fmaxf(rowmax[i], val);
      }
    }
#pragma unroll
    for (int i = 0; i < 8; ++i) {
#pragma unroll
      for (int off = 1; off < 16; off <<= 1)
        rowmax[i] = fmaxf(rowmax[i], __shfl_xor(rowmax[i], off));
    }

    // Online softmax update
    float rs[8];
#pragma unroll
    for (int i = 0; i < 8; ++i) {
      const float mnew  = fmaxf(m_i[i], rowmax[i]);
      const float alpha = exp2f(m_i[i] - mnew);
      m_i[i] = mnew;
      l_i[i] *= alpha;
#pragma unroll
      for (int dt = 0; dt < 4; ++dt) o[dt][i] *= alpha;
      rs[i] = 0.0f;
    }
#pragma unroll
    for (int nt = 0; nt < 4; ++nt) {
#pragma unroll
      for (int i = 0; i < 8; ++i) {
        const float p = exp2f(s[nt][i] - m_i[i]);
        s[nt][i] = p;
        rs[i] += p;
      }
    }
#pragma unroll
    for (int i = 0; i < 8; ++i) {
#pragma unroll
      for (int off = 1; off < 16; off <<= 1) rs[i] += __shfl_xor(rs[i], off);
      l_i[i] += rs[i];
    }

    // Re-layout P (C layout -> A layout) through per-wave LDS
#pragma unroll
    for (int nt = 0; nt < 4; ++nt)
#pragma unroll
      for (int i = 0; i < 8; ++i)
        pT[wid][i + laneHalf * 8][nt * 16 + lane16] = f2bf(s[nt][i]);

    FragB pf[2];
#pragma unroll
    for (int kh2 = 0; kh2 < 2; ++kh2) {
      pf[kh2].q[0] = *(const uint4*)&pT[wid][lane16][kh2 * 32 + laneHalf * 8];
      pf[kh2].q[1] = *(const uint4*)&pT[wid][lane16][kh2 * 32 + 16 + laneHalf * 8];
    }

    // O += P V   (B frags from transposed V in LDS: interleaved k chunks as A)
#pragma unroll
    for (int dt = 0; dt < 4; ++dt) {
#pragma unroll
      for (int kh2 = 0; kh2 < 2; ++kh2) {
        FragB vf;
        vf.q[0] = *(const uint4*)&vT[dt * 16 + lane16][kh2 * 32 + laneHalf * 8];
        vf.q[1] = *(const uint4*)&vT[dt * 16 + lane16][kh2 * 32 + 16 + laneHalf * 8];
        o[dt] = wmma_bf16(pf[kh2], vf, o[dt]);
      }
    }
    __syncthreads();
  }

  // Epilogue: y[b][t][h*64+d] = O / l
  const int bb = bh >> 4;
  const int hh = bh & 15;
  float inv[8];
#pragma unroll
  for (int i = 0; i < 8; ++i) inv[i] = 1.0f / l_i[i];
#pragma unroll
  for (int dt = 0; dt < 4; ++dt) {
#pragma unroll
    for (int i = 0; i < 8; ++i) {
      const int t = q0 + wid * 16 + i + laneHalf * 8;
      const int c = hh * 64 + dt * 16 + lane16;
      out[((size_t)(bb * T_DIM + t)) * E_DIM + c] = o[dt][i] * inv[i];
    }
  }
}

// ===========================================================================
extern "C" void kernel_launch(void* const* d_in, const int* in_sizes, int n_in,
                              void* d_out, int out_size, void* d_ws, size_t ws_size,
                              hipStream_t stream) {
  (void)in_sizes; (void)n_in; (void)out_size; (void)ws_size;
  const float* x    = (const float*)d_in[0];
  const float* w    = (const float*)d_in[1];
  const float* bias = (const float*)d_in[2];
  float* out = (float*)d_out;

  u16* qkv = (u16*)d_ws;                         // 3 x 16 MB bf16 [B,H,T,D]
  u16* xb  = qkv + 3 * (size_t)QK_ELEMS;         // 16 MB bf16 x
  u16* wb  = xb + (size_t)X_ELEMS;               // 6 MB bf16 w (3E live rows)

  cvt_bf16_kernel<<<X_ELEMS / 8 / 256, 256, 0, stream>>>(x, xb, X_ELEMS / 8);
  cvt_bf16_kernel<<<W_ELEMS / 8 / 256, 256, 0, stream>>>(w, wb, W_ELEMS / 8);

  dim3 g1(3 * E_DIM / GE_BN, (B_DIM * T_DIM) / GE_BM);  // 24 x 64
  qkv_gemm_kernel<<<g1, 256, 0, stream>>>(xb, wb, bias, qkv);

  dim3 g2(T_DIM / 64, B_DIM * H_DIM);  // 32 x 64
  flash_attn_kernel<<<g2, 128, 0, stream>>>(qkv, qkv + QK_ELEMS,
                                            qkv + 2 * (size_t)QK_ELEMS, out);
}
